// CFDFVnewGCN_86122684219979
// MI455X (gfx1250) — matrix operand order repaired and checked by reading, not compiled, not run on hardware
//
#include <hip/hip_runtime.h>
#include <hip/hip_bf16.h>

// ---------------------------------------------------------------------------
// CFD FVnewGCN on gfx1250: 6 x (edge-conditioned message + scatter-add +
// WMMA node GEMM with fused tanh/relu epilogue).
// ---------------------------------------------------------------------------

#define N_NODES   50000
#define N_EDGES   200000
#define XCS       520          // xc row stride (floats), >= max in_c (516)
#define KPAD_MAX  1568         // ceil(3*516/32)*32
#define MTILES    (N_NODES/16) // 3125 (N divisible by 16)

typedef __attribute__((ext_vector_type(16))) _Float16 v16h;
typedef __attribute__((ext_vector_type(8)))  float    v8f;

// ---------------- utility kernels ----------------

__global__ __launch_bounds__(256) void zero_f4(float4* __restrict__ p, long long n4) {
    long long i = (long long)blockIdx.x * blockDim.x + threadIdx.x;
    long long stride = (long long)gridDim.x * blockDim.x;
    float4 z; z.x = 0.f; z.y = 0.f; z.z = 0.f; z.w = 0.f;
    for (; i < n4; i += stride) p[i] = z;
}

__global__ __launch_bounds__(256) void prep0_kernel(const float* __restrict__ x,
                                                    const float* __restrict__ sdf,
                                                    const float* __restrict__ na,
                                                    float* __restrict__ xc) {
    int n = blockIdx.x * blockDim.x + threadIdx.x;
    if (n >= N_NODES) return;
    float* r = xc + (size_t)n * XCS;
    const float* xr = x + (size_t)n * 5;
    r[0] = xr[0]; r[1] = xr[1]; r[2] = xr[2]; r[3] = xr[3]; r[4] = xr[4];
    r[5] = sdf[n];
    r[6] = na[n];
}

__global__ __launch_bounds__(256) void fill_col_kernel(float* __restrict__ buf,
                                                       const float* __restrict__ src,
                                                       int col) {
    int n = blockIdx.x * blockDim.x + threadIdx.x;
    if (n >= N_NODES) return;
    buf[(size_t)n * XCS + col] = src[n];
}

__global__ __launch_bounds__(256) void fill_y_kernel(float* __restrict__ buf,
                                                     const float* __restrict__ y) {
    int idx = blockIdx.x * blockDim.x + threadIdx.x;
    if (idx >= N_NODES * 3) return;
    int n = idx / 3, c = idx % 3;
    buf[(size_t)n * XCS + c] = y[idx];
}

// pad+convert wout [outc, K] f32 -> wp [outcPad, Kpad] f16 (zero padded)
__global__ __launch_bounds__(256) void pack_w_kernel(const float* __restrict__ w,
                                                     _Float16* __restrict__ wp,
                                                     int outc, int K, int Kpad,
                                                     long long total) {
    long long i = (long long)blockIdx.x * blockDim.x + threadIdx.x;
    if (i >= total) return;
    int n = (int)(i / Kpad);
    int k = (int)(i % Kpad);
    float v = (n < outc && k < K) ? w[(size_t)n * K + k] : 0.0f;
    wp[i] = (_Float16)v;
}

// ---------------- edge kernel: scaling = relu(edge_attr @ win.T + bin);
// aggr[dst, j] += scaling[j] * xc[src, j/3]  (one block per edge) ----------

__global__ __launch_bounds__(256) void edge_kernel(const float* __restrict__ edge_attr,
                                                   const int* __restrict__ ei,
                                                   const float* __restrict__ win,
                                                   const float* __restrict__ bin,
                                                   const float* __restrict__ xc,
                                                   float* __restrict__ aggr,
                                                   int inc, int kpad) {
    int e = blockIdx.x;
    int s = ei[e];
    int d = ei[N_EDGES + e];
    const float* ea = edge_attr + (size_t)e * 6;
    float e0 = ea[0], e1 = ea[1], e2 = ea[2], e3 = ea[3], e4 = ea[4], e5 = ea[5];
    const float* xr = xc + (size_t)s * XCS;
    float* ar = aggr + (size_t)d * kpad;
    int K = 3 * inc;
    for (int j = threadIdx.x; j < K; j += blockDim.x) {
        const float* w = win + (size_t)j * 6;
        float sc = w[0]*e0 + w[1]*e1 + w[2]*e2 + w[3]*e3 + w[4]*e4 + w[5]*e5 + bin[j];
        sc = fmaxf(sc, 0.0f);
        atomicAdd(&ar[j], sc * xr[j / 3]);
    }
}

// ---------------- WMMA node GEMM: dst = act(aggr[N,Kpad] @ wp.T + bout) ----
// One wave computes a 16-row strip over NT consecutive 16-col tiles.
// A fragment (16x32 f16): lane L -> row=L%16, k = kk + (L/16)*16 + {0..15}
// B fragment (32x16 f16): lane L -> col=L%16, k = kk + (L/16)*16 + {0..15}
// C/D (16x16 f32, v8f): VGPR r -> M = r + 8*(L/16), N = L%16

template <int NT>
__global__ __launch_bounds__(256) void gemm_tanh_kernel(const float* __restrict__ aggr,
                                                        const _Float16* __restrict__ wp,
                                                        const float* __restrict__ bout,
                                                        float* __restrict__ dst,
                                                        int Kpad, int outc, int nGroups,
                                                        int dstStride, int col0, int doRelu) {
    int wave = (blockIdx.x * blockDim.x + threadIdx.x) >> 5;
    int lane = threadIdx.x & 31;
    int totalWaves = MTILES * nGroups;
    if (wave >= totalWaves) return;
    int tm = wave / nGroups;
    int tg = wave % nGroups;
    int nbase = tg * NT * 16;
    int laneLo = lane & 15;
    int laneHi = lane >> 4;

    v8f acc[NT] = {};

    int row = tm * 16 + laneLo;
    const float* arow = aggr + (size_t)row * Kpad;

    for (int kk = 0; kk < Kpad; kk += 32) {
        int ka = kk + laneHi * 16;
        float4 a0 = *(const float4*)(arow + ka);
        float4 a1 = *(const float4*)(arow + ka + 4);
        float4 a2 = *(const float4*)(arow + ka + 8);
        float4 a3 = *(const float4*)(arow + ka + 12);
        v16h A = {};
        A[0]  = (_Float16)a0.x; A[1]  = (_Float16)a0.y; A[2]  = (_Float16)a0.z; A[3]  = (_Float16)a0.w;
        A[4]  = (_Float16)a1.x; A[5]  = (_Float16)a1.y; A[6]  = (_Float16)a1.z; A[7]  = (_Float16)a1.w;
        A[8]  = (_Float16)a2.x; A[9]  = (_Float16)a2.y; A[10] = (_Float16)a2.z; A[11] = (_Float16)a2.w;
        A[12] = (_Float16)a3.x; A[13] = (_Float16)a3.y; A[14] = (_Float16)a3.z; A[15] = (_Float16)a3.w;

        int kb = kk + laneHi * 16;
#pragma unroll
        for (int t = 0; t < NT; ++t) {
            int col = nbase + t * 16 + laneLo;
            v16h B = *(const v16h*)(wp + (size_t)col * Kpad + kb);
            acc[t] = __builtin_amdgcn_wmma_f32_16x16x32_f16(
                false, A, false, B, (short)0, acc[t], false, false);
        }
    }

    int mbase = tm * 16 + laneHi * 8;
#pragma unroll
    for (int t = 0; t < NT; ++t) {
        int n = nbase + t * 16 + laneLo;
        if (n >= outc) continue;
        float bias = bout[n];
#pragma unroll
        for (int r = 0; r < 8; ++r) {
            float v = tanhf(acc[t][r] + bias);
            if (doRelu) v = fmaxf(v, 0.0f);
            dst[(size_t)(mbase + r) * dstStride + col0 + n] = v;
        }
    }
}

// ---------------- host orchestration ----------------

extern "C" void kernel_launch(void* const* d_in, const int* in_sizes, int n_in,
                              void* d_out, int out_size, void* d_ws, size_t ws_size,
                              hipStream_t stream) {
    (void)in_sizes; (void)n_in; (void)out_size; (void)ws_size;

    const float* x         = (const float*)d_in[0];
    const float* sdf       = (const float*)d_in[1];
    const float* na        = (const float*)d_in[2];
    const float* edge_attr = (const float*)d_in[3];
    const float* yorig     = (const float*)d_in[4];
    const int*   ei        = (const int*)d_in[5];

    const float *winp[6], *binp[6], *woutp[6], *boutp[6];
    for (int l = 0; l < 6; ++l) {
        winp[l]  = (const float*)d_in[6 + 4*l + 0];
        binp[l]  = (const float*)d_in[6 + 4*l + 1];
        woutp[l] = (const float*)d_in[6 + 4*l + 2];
        boutp[l] = (const float*)d_in[6 + 4*l + 3];
    }

    char* ws = (char*)d_ws;
    float*    XA = (float*)ws;                                   // N*520 f32
    float*    XB = (float*)(ws + 104000000);                     // N*520 f32
    float*    AG = (float*)(ws + 208000000);                     // N*1568 f32
    _Float16* WP = (_Float16*)(ws + 208000000 + 313600000);      // 512*1568 f16

    static const int LINC[6] = {7, 513, 513, 516, 513, 513};

    auto layer = [&](int l, const float* xcIn, float* dst, int dstStride, int col0, int relu) {
        int inc  = LINC[l];
        int K    = 3 * inc;
        int kpad = (K + 31) & ~31;
        int outc = (l == 5) ? 3 : 512;
        int outcPad = (l == 5) ? 16 : 512;

        long long n4 = (long long)N_NODES * kpad / 4;
        zero_f4<<<4096, 256, 0, stream>>>((float4*)AG, n4);

        edge_kernel<<<N_EDGES, 256, 0, stream>>>(edge_attr, ei, winp[l], binp[l],
                                                 xcIn, AG, inc, kpad);

        long long wtot = (long long)outcPad * kpad;
        pack_w_kernel<<<(int)((wtot + 255) / 256), 256, 0, stream>>>(
            woutp[l], WP, outc, K, kpad, wtot);

        int ntN = outcPad / 16;
        if (l == 5) {
            int groups = ntN;                         // NT = 1
            int blocks = (MTILES * groups * 32 + 255) / 256;
            gemm_tanh_kernel<1><<<blocks, 256, 0, stream>>>(
                AG, WP, boutp[l], dst, kpad, outc, groups, dstStride, col0, relu);
        } else {
            int groups = ntN / 8;                     // NT = 8
            int blocks = (MTILES * groups * 32 + 255) / 256;
            gemm_tanh_kernel<8><<<blocks, 256, 0, stream>>>(
                AG, WP, boutp[l], dst, kpad, outc, groups, dstStride, col0, relu);
        }
    };

    int nb = (N_NODES + 255) / 256;

    // layer p0: xc = [x(5), sdf(1), node_attr(1)]
    prep0_kernel<<<nb, 256, 0, stream>>>(x, sdf, na, XA);
    layer(0, XA, XB, XCS, 0, 1);
    fill_col_kernel<<<nb, 256, 0, stream>>>(XB, na, 512);

    // layer p1
    layer(1, XB, XA, XCS, 0, 1);
    fill_col_kernel<<<nb, 256, 0, stream>>>(XA, na, 512);

    // layer p2 -> fine_x written at col 3 of XB to form c0 input
    layer(2, XA, XB, XCS, 3, 1);
    fill_y_kernel<<<(N_NODES * 3 + 255) / 256, 256, 0, stream>>>(XB, yorig);
    fill_col_kernel<<<nb, 256, 0, stream>>>(XB, na, 515);

    // layer c0 (in_c = 516)
    layer(3, XB, XA, XCS, 0, 1);
    fill_col_kernel<<<nb, 256, 0, stream>>>(XA, na, 512);

    // layer c1
    layer(4, XA, XB, XCS, 0, 1);
    fill_col_kernel<<<nb, 256, 0, stream>>>(XB, na, 512);

    // layer c2: out_c = 3, tanh only, straight to d_out
    layer(5, XB, (float*)d_out, 3, 0, 0);
}
